// SeSaMe_75677323755600
// MI455X (gfx1250) — matrix operand order repaired and checked by reading, not compiled, not run on hardware
//
#include <hip/hip_runtime.h>
#include <hip/hip_bf16.h>

// ---------------------------------------------------------------------------
// CDNA5 (gfx1250) implementation of the hierarchical Mamba reference.
// GEMMs run on v_wmma_f32_16x16x32_bf16 (f32 accumulate, bf16 operands staged
// through LDS with b128 loads). Edge tiles use clamped vector loads + register
// selects (fully branch-free). Activation is a template parameter so each
// instantiation carries exactly one epilogue path.
// ---------------------------------------------------------------------------

typedef __attribute__((ext_vector_type(16))) __bf16 v16bf;
typedef __attribute__((ext_vector_type(8)))  __bf16 bf16x8;
typedef __attribute__((ext_vector_type(8)))  float  v8f;
typedef __attribute__((ext_vector_type(4)))  float  f4;

#define TBM 128
#define TBN 64
#define TBK 32
#define LDP (TBK + 8)   // 40 elems -> 80B pitch, rows 16B aligned, no bank conflicts

__device__ __forceinline__ unsigned short bfbits(float f) {
  unsigned u = __builtin_bit_cast(unsigned, f);
  u += 0x7FFFu + ((u >> 16) & 1u);            // round-to-nearest-even
  return (unsigned short)(u >> 16);
}
__device__ __forceinline__ unsigned pack2(float a, float b) {
  return (unsigned)bfbits(a) | ((unsigned)bfbits(b) << 16);
}
__device__ __forceinline__ float selz(float v, bool ok) { return ok ? v : 0.0f; }

__device__ __forceinline__ float gelu_f(float x) {
  return 0.5f * x * (1.0f + erff(x * 0.70710678118654752f));
}
__device__ __forceinline__ float silu_f(float x) {
  return x / (1.0f + __expf(-x));
}
__device__ __forceinline__ float softplus_f(float x) {
  return (x > 20.0f) ? x : log1pf(__expf(x));
}
template <int ACT>
__device__ __forceinline__ float apply_act(float v) {
  if (ACT == 1) return gelu_f(v);
  if (ACT == 2) return silu_f(v);
  if (ACT == 3) return softplus_f(v);
  return v;
}

// ---------------------------------------------------------------------------
// Tiled WMMA GEMM: Y[M,N] = act(X[M,K](ldx) * W[N,K]^T + bias), Y has ldy.
// 256 threads = 8 waves arranged 4(M) x 2(N); each wave owns a 32x32 output
// (2x2 WMMA tiles -> 4 v_wmma per K-step, 2x fragment reuse each way).
// NOTE: requires ldx to be a multiple of 4 floats and >= ceil32(K) (true for
// every GEMM in this network), so edge-tile vector loads stay in-bounds.
// ---------------------------------------------------------------------------
template <int ACT>
__global__ __launch_bounds__(256)
void wmma_gemm_kernel(const float* __restrict__ X, int ldx,
                      const float* __restrict__ W,
                      const float* __restrict__ bias,
                      float* __restrict__ Y, int ldy,
                      int M, int N, int K) {
  __shared__ __bf16 As[TBM][LDP];
  __shared__ __bf16 Bs[TBN][LDP];

  const int tid  = threadIdx.x;
  const int lane = tid & 31;
  const int wave = tid >> 5;
  const int wm   = wave & 3;        // 0..3 : 32-row slab
  const int wn   = wave >> 2;       // 0..1 : 32-col slab
  const int blockRow = blockIdx.y * TBM;
  const int blockCol = blockIdx.x * TBN;

  v8f acc[2][2] = {};

  // staging assignment: A tile 128x32 -> thread = (row, 16-col seg)
  //                     B tile  64x32 -> thread = (row,  8-col seg)
  const int arow = tid >> 1;        // 0..127
  const int acs  = (tid & 1) * 16;  // 0,16
  const int brow = tid >> 2;        // 0..63
  const int bcs  = (tid & 3) * 8;   // 0,8,16,24

  const bool mFull = (blockRow + TBM <= M);
  const bool nFull = (blockCol + TBN <= N);

  for (int k0 = 0; k0 < K; k0 += TBK) {
    const bool kFull = (k0 + TBK <= K);
    if (mFull & nFull & kFull) {
      // -------- fast path: b128 loads, packed LDS stores --------
      const f4* xp = (const f4*)(X + (size_t)(blockRow + arow) * ldx + k0 + acs);
      f4 v0 = xp[0], v1 = xp[1], v2 = xp[2], v3 = xp[3];
      unsigned* da = (unsigned*)&As[arow][acs];
      da[0] = pack2(v0[0], v0[1]); da[1] = pack2(v0[2], v0[3]);
      da[2] = pack2(v1[0], v1[1]); da[3] = pack2(v1[2], v1[3]);
      da[4] = pack2(v2[0], v2[1]); da[5] = pack2(v2[2], v2[3]);
      da[6] = pack2(v3[0], v3[1]); da[7] = pack2(v3[2], v3[3]);

      const f4* wp = (const f4*)(W + (size_t)(blockCol + brow) * K + k0 + bcs);
      f4 w0 = wp[0], w1 = wp[1];
      unsigned* db = (unsigned*)&Bs[brow][bcs];
      db[0] = pack2(w0[0], w0[1]); db[1] = pack2(w0[2], w0[3]);
      db[2] = pack2(w1[0], w1[1]); db[3] = pack2(w1[2], w1[3]);

      if (k0 + TBK < K) {   // speculative prefetch of next K tile
        __builtin_prefetch(X + (size_t)(blockRow + arow) * ldx + k0 + TBK + acs, 0, 1);
        __builtin_prefetch(W + (size_t)(blockCol + brow) * K + k0 + TBK + bcs, 0, 1);
      }
    } else {
      // -------- edge path: clamped unconditional VECTOR loads, then register
      // selects. Loads are unconditional so nothing gets sunk under exec masks.
      const int gr    = blockRow + arow;
      const bool rOk  = gr < M;
      const int grc   = rOk ? gr : (M - 1);
      const f4* xp = (const f4*)(X + (size_t)grc * ldx + k0 + acs);
      f4 v0 = xp[0], v1 = xp[1], v2 = xp[2], v3 = xp[3];
      const int ka = k0 + acs;
      unsigned* da = (unsigned*)&As[arow][acs];
      da[0] = pack2(selz(v0[0], rOk && ka + 0  < K), selz(v0[1], rOk && ka + 1  < K));
      da[1] = pack2(selz(v0[2], rOk && ka + 2  < K), selz(v0[3], rOk && ka + 3  < K));
      da[2] = pack2(selz(v1[0], rOk && ka + 4  < K), selz(v1[1], rOk && ka + 5  < K));
      da[3] = pack2(selz(v1[2], rOk && ka + 6  < K), selz(v1[3], rOk && ka + 7  < K));
      da[4] = pack2(selz(v2[0], rOk && ka + 8  < K), selz(v2[1], rOk && ka + 9  < K));
      da[5] = pack2(selz(v2[2], rOk && ka + 10 < K), selz(v2[3], rOk && ka + 11 < K));
      da[6] = pack2(selz(v3[0], rOk && ka + 12 < K), selz(v3[1], rOk && ka + 13 < K));
      da[7] = pack2(selz(v3[2], rOk && ka + 14 < K), selz(v3[3], rOk && ka + 15 < K));

      const int gc    = blockCol + brow;
      const bool cOk  = gc < N;
      const int gcc   = cOk ? gc : (N - 1);
      // clamp the flat index so the 8-float vector load stays inside W[N*K];
      // clamped chunks are always fully zeroed by the selects below.
      size_t base = (size_t)gcc * K + k0 + bcs;
      size_t maxb = (size_t)N * K;
      maxb = (maxb >= 8) ? (maxb - 8) : 0;
      if (base > maxb) base = maxb;
      base &= ~(size_t)3;                       // keep 16B alignment
      const f4* wp = (const f4*)(W + base);
      f4 w0 = wp[0], w1 = wp[1];
      const int kbq = k0 + bcs;
      unsigned* db = (unsigned*)&Bs[brow][bcs];
      db[0] = pack2(selz(w0[0], cOk && kbq + 0 < K), selz(w0[1], cOk && kbq + 1 < K));
      db[1] = pack2(selz(w0[2], cOk && kbq + 2 < K), selz(w0[3], cOk && kbq + 3 < K));
      db[2] = pack2(selz(w1[0], cOk && kbq + 4 < K), selz(w1[1], cOk && kbq + 5 < K));
      db[3] = pack2(selz(w1[2], cOk && kbq + 6 < K), selz(w1[3], cOk && kbq + 7 < K));
    }
    __syncthreads();

    // ---- A fragments: 16x32 bf16 (M = lane%16; K groups of 8 per lane half)
    const int ar = wm * 32 + (lane & 15);
    const int kb = (lane >> 4) * 8;
    v16bf af0, af1;
    {
      bf16x8 lo  = *(const bf16x8*)&As[ar][kb];
      bf16x8 hi  = *(const bf16x8*)&As[ar][kb + 16];
      bf16x8 lo1 = *(const bf16x8*)&As[ar + 16][kb];
      bf16x8 hi1 = *(const bf16x8*)&As[ar + 16][kb + 16];
      #pragma unroll
      for (int i = 0; i < 8; ++i) {
        af0[i] = lo[i];  af0[8 + i] = hi[i];
        af1[i] = lo1[i]; af1[8 + i] = hi1[i];
      }
    }
    // ---- B fragments: 32x16 bf16 (N = lane%16; K groups of 8 per lane half)
    const int bc = wn * 32 + (lane & 15);
    const int kh = (lane >> 4) * 8;
    v16bf bv0, bv1;
    {
      bf16x8 p0 = *(const bf16x8*)&Bs[bc][kh];
      bf16x8 p1 = *(const bf16x8*)&Bs[bc][kh + 16];
      bf16x8 q0 = *(const bf16x8*)&Bs[bc + 16][kh];
      bf16x8 q1 = *(const bf16x8*)&Bs[bc + 16][kh + 16];
      #pragma unroll
      for (int i = 0; i < 8; ++i) {
        bv0[i] = p0[i]; bv0[8 + i] = p1[i];
        bv1[i] = q0[i]; bv1[8 + i] = q1[i];
      }
    }

    acc[0][0] = __builtin_amdgcn_wmma_f32_16x16x32_bf16(false, af0, false, bv0,
                                                        (short)0, acc[0][0], false, false);
    acc[0][1] = __builtin_amdgcn_wmma_f32_16x16x32_bf16(false, af0, false, bv1,
                                                        (short)0, acc[0][1], false, false);
    acc[1][0] = __builtin_amdgcn_wmma_f32_16x16x32_bf16(false, af1, false, bv0,
                                                        (short)0, acc[1][0], false, false);
    acc[1][1] = __builtin_amdgcn_wmma_f32_16x16x32_bf16(false, af1, false, bv1,
                                                        (short)0, acc[1][1], false, false);
    __syncthreads();
  }

  // ---- epilogue: VGPR r -> M = r (lanes 0-15) / 8+r (lanes 16-31)
  const int half = lane >> 4;
  const int ln   = lane & 15;
  const int gc0  = blockCol + wn * 32 + ln;
  const int gc1  = gc0 + 16;
  // hoisted, index-clamped bias loads (value unused when column is OOB)
  const float b0 = bias ? bias[gc0 < N ? gc0 : (N - 1)] : 0.0f;
  const float b1 = bias ? bias[gc1 < N ? gc1 : (N - 1)] : 0.0f;

  if (mFull & nFull) {
    #pragma unroll
    for (int mi = 0; mi < 2; ++mi) {
      #pragma unroll
      for (int r = 0; r < 8; ++r) {
        const int grow = blockRow + wm * 32 + mi * 16 + half * 8 + r;
        Y[(size_t)grow * ldy + gc0] = apply_act<ACT>(acc[mi][0][r] + b0);
        Y[(size_t)grow * ldy + gc1] = apply_act<ACT>(acc[mi][1][r] + b1);
      }
    }
  } else {
    #pragma unroll
    for (int mi = 0; mi < 2; ++mi) {
      #pragma unroll
      for (int r = 0; r < 8; ++r) {
        const int grow = blockRow + wm * 32 + mi * 16 + half * 8 + r;
        if (grow >= M) continue;
        if (gc0 < N) Y[(size_t)grow * ldy + gc0] = apply_act<ACT>(acc[mi][0][r] + b0);
        if (gc1 < N) Y[(size_t)grow * ldy + gc1] = apply_act<ACT>(acc[mi][1][r] + b1);
      }
    }
  }
}

// ---------------------------------------------------------------------------
// Depthwise causal conv (width 4) + SiLU. xi is a column slice of xz.
// ---------------------------------------------------------------------------
__global__ void conv_silu_kernel(const float* __restrict__ xi, int ldxi,
                                 const float* __restrict__ cw,
                                 const float* __restrict__ cb,
                                 float* __restrict__ xc,
                                 int Bb, int Lm, int di) {
  const long long gid = (long long)blockIdx.x * blockDim.x + threadIdx.x;
  const long long tot = (long long)Bb * Lm * di;
  if (gid >= tot) return;
  const int c = (int)(gid % di);
  const int t = (int)((gid / di) % Lm);
  const int b = (int)(gid / ((long long)di * Lm));
  float acc = cb[c];
  #pragma unroll
  for (int j = 0; j < 4; ++j) {
    const int ts = t - 3 + j;
    if (ts >= 0)
      acc += xi[(size_t)(b * Lm + ts) * ldxi + c] * cw[c * 4 + j];
  }
  xc[gid] = silu_f(acc);
}

// ---------------------------------------------------------------------------
// Selective scan, one lane per (batch, channel). 16 states in registers.
// Fuses the (y + xc*D) * silu(z) gating so out-proj reads one tensor.
// ---------------------------------------------------------------------------
__global__ void scan_gate_kernel(const float* __restrict__ delta,  // (M,di)
                                 const float* __restrict__ xc,     // (M,di)
                                 const float* __restrict__ Bp,     // (M,*) ldbc
                                 const float* __restrict__ Cp,     // (M,*) ldbc
                                 int ldbc,
                                 const float* __restrict__ zp,     // (M,*) ldz
                                 int ldz,
                                 const float* __restrict__ Alog,   // (di,16)
                                 const float* __restrict__ Dv,     // (di)
                                 float* __restrict__ yg,           // (M,di)
                                 int Bb, int Lm, int di) {
  const int gid = blockIdx.x * blockDim.x + threadIdx.x;
  if (gid >= Bb * di) return;
  const int b = gid / di;
  const int d = gid % di;

  float A[16], h[16];
  #pragma unroll
  for (int n = 0; n < 16; ++n) {
    A[n] = -__expf(Alog[(size_t)d * 16 + n]);
    h[n] = 0.0f;
  }
  const float Dd = Dv[d];

  for (int t = 0; t < Lm; ++t) {
    const size_t row = (size_t)b * Lm + t;
    const float dl = delta[row * di + d];
    const float xv = xc[row * di + d];
    const float dx = dl * xv;
    float y = 0.0f;
    #pragma unroll
    for (int n = 0; n < 16; ++n) {
      const float dA = __expf(dl * A[n]);
      h[n] = dA * h[n] + dx * Bp[row * ldbc + n];
      y += h[n] * Cp[row * ldbc + n];
    }
    const float zv = zp[row * ldz + d];
    yg[row * di + d] = (y + xv * Dd) * silu_f(zv);
  }
}

__global__ void ew_add_kernel(const float* __restrict__ a,
                              const float* __restrict__ b,
                              float* __restrict__ o, long long n) {
  const long long i = (long long)blockIdx.x * blockDim.x + threadIdx.x;
  if (i < n) o[i] = a[i] + b[i];
}

// ---------------------------------------------------------------------------
// Host-side helpers
// ---------------------------------------------------------------------------
static inline void gemm(hipStream_t s, const float* X, int ldx, const float* W,
                        const float* b, float* Y, int ldy,
                        int M, int N, int K, int act) {
  dim3 grid((N + TBN - 1) / TBN, (M + TBM - 1) / TBM);
  if (act == 1)
    wmma_gemm_kernel<1><<<grid, 256, 0, s>>>(X, ldx, W, b, Y, ldy, M, N, K);
  else if (act == 3)
    wmma_gemm_kernel<3><<<grid, 256, 0, s>>>(X, ldx, W, b, Y, ldy, M, N, K);
  else
    wmma_gemm_kernel<0><<<grid, 256, 0, s>>>(X, ldx, W, b, Y, ldy, M, N, K);
}

struct MambaW {
  const float *in_w, *conv_w, *conv_b, *xproj_w, *dt_w, *dt_b, *Alog, *Dv, *out_w;
};

static inline void run_mamba(hipStream_t s, const float* u, int Bb, int Lm, int dm,
                             const MambaW& w,
                             float* xz, float* xc, float* dbc, float* delta,
                             float* yg, float* outY, int outAct) {
  const int di   = 2 * dm;
  const int dtr  = (dm / 16) > 0 ? (dm / 16) : 1;
  const int M    = Bb * Lm;
  const int ldbc = dtr + 32;           // [dt | B(16) | C(16)]

  // in-proj: xz = u @ in_w^T    (M x 2di)
  gemm(s, u, dm, w.in_w, nullptr, xz, 2 * di, M, 2 * di, dm, 0);
  // depthwise conv + silu on xi = xz[:, :di]
  {
    long long tot = (long long)M * di;
    conv_silu_kernel<<<(unsigned)((tot + 255) / 256), 256, 0, s>>>(
        xz, 2 * di, w.conv_w, w.conv_b, xc, Bb, Lm, di);
  }
  // x-proj: dbc = xc @ xproj^T  (M x (dtr+32))
  gemm(s, xc, di, w.xproj_w, nullptr, dbc, ldbc, M, ldbc, di, 0);
  // delta = softplus(dt @ dt_w^T + dt_b)   (M x di)
  gemm(s, dbc, ldbc, w.dt_w, w.dt_b, delta, di, M, di, dtr, 3);
  // scan + gate
  {
    int nth = Bb * di;
    scan_gate_kernel<<<(nth + 255) / 256, 256, 0, s>>>(
        delta, xc, dbc + dtr, dbc + dtr + 16, ldbc,
        xz + di, 2 * di, w.Alog, w.Dv, yg, Bb, Lm, di);
  }
  // out-proj (optionally fused GELU)
  gemm(s, yg, di, w.out_w, nullptr, outY, dm, M, dm, di, outAct);
}

extern "C" void kernel_launch(void* const* d_in, const int* in_sizes, int n_in,
                              void* d_out, int out_size, void* d_ws, size_t ws_size,
                              hipStream_t stream) {
  (void)in_sizes; (void)n_in; (void)out_size; (void)ws_size;
  const int B = 2, L = 4096, H = 128;

  const float* x    = (const float*)d_in[0];
  const float* l1_w = (const float*)d_in[1];
  const float* l1_b = (const float*)d_in[2];
  const float* l2_w = (const float*)d_in[3];
  const float* l2_b = (const float*)d_in[4];
  const float* l3_w = (const float*)d_in[5];
  const float* l3_b = (const float*)d_in[6];
  const float* l4_w = (const float*)d_in[7];
  const float* l4_b = (const float*)d_in[8];
  MambaW m1{(const float*)d_in[9],  (const float*)d_in[10], (const float*)d_in[11],
            (const float*)d_in[12], (const float*)d_in[13], (const float*)d_in[14],
            (const float*)d_in[15], (const float*)d_in[16], (const float*)d_in[17]};
  MambaW m2{(const float*)d_in[18], (const float*)d_in[19], (const float*)d_in[20],
            (const float*)d_in[21], (const float*)d_in[22], (const float*)d_in[23],
            (const float*)d_in[24], (const float*)d_in[25], (const float*)d_in[26]};
  MambaW m3{(const float*)d_in[27], (const float*)d_in[28], (const float*)d_in[29],
            (const float*)d_in[30], (const float*)d_in[31], (const float*)d_in[32],
            (const float*)d_in[33], (const float*)d_in[34], (const float*)d_in[35]};

  float* ws = (float*)d_ws;
  size_t off = 0;
  auto alloc = [&](size_t n) { float* p = ws + off; off += n; return p; };

  float* x1     = alloc((size_t)B * 1024 * 256);
  float* x2     = alloc((size_t)B * 256 * 512);
  float* x3     = alloc((size_t)B * 256 * 512);
  float* x4     = alloc((size_t)B * 256 * 1024);
  float* s2     = alloc((size_t)B * 1024 * 256);
  float* x6     = alloc((size_t)B * 1024 * 512);
  float* s3     = alloc((size_t)B * 4096 * 128);
  // mamba1 scratch (M=512, di=1024, dtr=32)
  float* xz1    = alloc((size_t)512 * 2048);
  float* xc1    = alloc((size_t)512 * 1024);
  float* dbc1   = alloc((size_t)512 * 64);
  float* dl1    = alloc((size_t)512 * 1024);
  float* yg1    = alloc((size_t)512 * 1024);
  float* mo1    = alloc((size_t)512 * 512);
  // mamba2 scratch (M=2048, di=512, dtr=16)
  float* xz2    = alloc((size_t)2048 * 1024);
  float* xc2    = alloc((size_t)2048 * 512);
  float* dbc2   = alloc((size_t)2048 * 48);
  float* dl2    = alloc((size_t)2048 * 512);
  float* yg2    = alloc((size_t)2048 * 512);
  float* x5     = alloc((size_t)2048 * 256);
  // mamba3 scratch (M=8192, di=256, dtr=8)
  float* xz3    = alloc((size_t)8192 * 512);
  float* xc3    = alloc((size_t)8192 * 256);
  float* dbc3   = alloc((size_t)8192 * 40);
  float* dl3    = alloc((size_t)8192 * 256);
  float* yg3    = alloc((size_t)8192 * 256);

  // ---- stage 1: x1 = gelu(x0 @ l1^T + b)    (2048 x 256, K=512)
  gemm(stream, x, 4 * H, l1_w, l1_b, x1, 2 * H, B * (L / 4), 2 * H, 4 * H, 1);
  // ---- stage 2: x2 = gelu(x1r @ l2^T + b)   (512 x 512, K=1024)
  gemm(stream, x1, 8 * H, l2_w, l2_b, x2, 4 * H, B * (L / 16), 4 * H, 8 * H, 1);

  // ---- mamba1, x3 = x2 + gelu(mamba1(x2))
  run_mamba(stream, x2, B, L / 16, 4 * H, m1, xz1, xc1, dbc1, dl1, yg1, mo1, /*gelu*/1);
  {
    long long n = (long long)B * 256 * 512;
    ew_add_kernel<<<(unsigned)((n + 255) / 256), 256, 0, stream>>>(x2, mo1, x3, n);
  }

  // ---- stage 4: x4 = gelu(x3 @ l3^T + b)    (512 x 1024, K=512)
  gemm(stream, x3, 4 * H, l3_w, l3_b, x4, 8 * H, B * (L / 16), 8 * H, 4 * H, 1);
  // s2 = x4(reshaped) + x1
  {
    long long n = (long long)B * 1024 * 256;
    ew_add_kernel<<<(unsigned)((n + 255) / 256), 256, 0, stream>>>(x4, x1, s2, n);
  }

  // ---- mamba2, x5 = gelu(mamba2(s2))  (gelu fused into out-proj)
  run_mamba(stream, s2, B, L / 4, 2 * H, m2, xz2, xc2, dbc2, dl2, yg2, x5, /*gelu*/1);

  // ---- stage 6: x6 = gelu(x5 @ l4^T + b)    (2048 x 512, K=256)
  gemm(stream, x5, 2 * H, l4_w, l4_b, x6, 4 * H, B * (L / 4), 4 * H, 2 * H, 1);
  // s3 = x6(reshaped) + x
  {
    long long n = (long long)B * L * H;
    ew_add_kernel<<<(unsigned)((n + 255) / 256), 256, 0, stream>>>(x6, x, s3, n);
  }

  // ---- mamba3, out = gelu(mamba3(s3)) written straight to d_out
  run_mamba(stream, s3, B, L, H, m3, xz3, xc3, dbc3, dl3, yg3, (float*)d_out, /*gelu*/1);
}
